// WeightedCrossAttention_13804024889564
// MI455X (gfx1250) — matrix operand order repaired and checked by reading, not compiled, not run on hardware
//
#include <hip/hip_runtime.h>

// ---------------- problem constants ----------------
#define D_    256
#define HW_   16384
#define B_    8
#define N_    256
#define S_    256
#define NH_   8
#define DH_   32
#define IMP_  192
#define COV_  64
#define H2_   128
#define EPS_  1e-5f
#define QSCALE 0.17677669529663687f   // 1/sqrt(32)

typedef __attribute__((ext_vector_type(16))) _Float16 v16h;
typedef __attribute__((ext_vector_type(8)))  _Float16 v8h;
typedef __attribute__((ext_vector_type(8)))  float    v8f;

// =====================================================================
// K1: full bitonic argsort (descending value, ascending index tiebreak)
// per row, entirely in LDS (dynamic, 128KB of the WGP's 320KB pool).
// =====================================================================
extern __shared__ unsigned char wca_dynlds[];

__global__ __launch_bounds__(1024) void wca_sort_select(
    const float* __restrict__ curio, const int* __restrict__ cov_sel,
    int* __restrict__ feat_idx)
{
  float* vals = (float*)wca_dynlds;
  int*   idxs = (int*)(wca_dynlds + HW_ * sizeof(float));
  const int n   = blockIdx.x;
  const int tid = threadIdx.x;
  const int nt  = blockDim.x;
  const float* row = curio + (size_t)n * HW_;
  for (int i = tid; i < HW_; i += nt) { vals[i] = row[i]; idxs[i] = i; }
  __syncthreads();
  for (int lk = 1; lk <= 14; ++lk) {          // k = 1<<lk, up to HW_
    const int k = 1 << lk;
    for (int js = lk - 1; js >= 0; --js) {
      const int j = 1 << js;
      for (int t = tid; t < (HW_ / 2); t += nt) {
        int i = ((t >> js) << (js + 1)) + (t & (j - 1));
        int p = i + j;
        bool up = ((i & k) != 0);             // ascending sub-sequence
        float va = vals[i], vb = vals[p];
        int   ia = idxs[i], ib = idxs[p];
        bool before = (va > vb) || (va == vb && ia < ib);  // descending order
        if (before == up) {
          vals[i] = vb; vals[p] = va;
          idxs[i] = ib; idxs[p] = ia;
        }
      }
      __syncthreads();
    }
  }
  for (int s = tid; s < IMP_; s += nt) feat_idx[n * S_ + s] = idxs[s];
  for (int t = tid; t < COV_; t += nt)
    feat_idx[n * S_ + IMP_ + t] = idxs[IMP_ + cov_sel[t]];
}

// =====================================================================
// K2: gather features/pos at selected indices; keys = f+p, values = f
// =====================================================================
__global__ __launch_bounds__(256) void wca_gather(
    const float* __restrict__ feats, const float* __restrict__ pos,
    const int* __restrict__ bidx, const int* __restrict__ fidx,
    _Float16* __restrict__ keys, _Float16* __restrict__ values)
{
  const int rowid = blockIdx.x;            // rowid = s*N + n
  const int s = rowid / N_, n = rowid % N_;
  const int i = fidx[n * S_ + s];
  const int b = bidx[n];
  const size_t base = ((size_t)i * B_ + b) * D_;
  const int d = threadIdx.x;
  const float f = feats[base + d];
  const float p = pos[base + d];
  keys[(size_t)rowid * D_ + d]   = (_Float16)(f + p);
  values[(size_t)rowid * D_ + d] = (_Float16)f;
}

// =====================================================================
// K3: WMMA GEMM  OUT[M,Nout] = act((A[M,K] @ W[Nout,K]^T + b)*scale)
// Compile-time variants (no runtime branching in hot loops).
// Block tile 64x128, 8 waves x (32x32), LDS K-stage 64 (2 WMMA k-chunks).
// All shapes are exact tile multiples -> no bounds checks.
// =====================================================================
#define BM 64
#define BN 128
#define BK 64
#define LDH (BK + 8)   // 72 halves = 144B row stride: 16B aligned

template <bool AF32, bool OF16, bool RELU>
__global__ __launch_bounds__(256) void wca_gemm_wmma(
    const void* __restrict__ Aptr,
    const float* __restrict__ W, const float* __restrict__ bias,
    void* __restrict__ Optr,
    int M, int K, int Nout, int ldOut, float scale)
{
  __shared__ _Float16 As[BM][LDH];
  __shared__ _Float16 Ws[BN][LDH];

  const int tid  = threadIdx.x;
  const int m0   = blockIdx.y * BM;
  const int n0   = blockIdx.x * BN;
  const int lane = tid & 31;
  const int wave = tid >> 5;
  const int wm   = (wave & 1) * 32;
  const int wn   = (wave >> 1) * 32;
  const int half = lane >> 4;

  const float*    Af = (const float*)Aptr;
  const _Float16* Ah = (const _Float16*)Aptr;

  v8f acc[2][2];
  #pragma unroll
  for (int i2 = 0; i2 < 2; ++i2)
    #pragma unroll
    for (int j2 = 0; j2 < 2; ++j2) {
      float bv = bias[n0 + wn + j2 * 16 + (lane & 15)];
      #pragma unroll
      for (int r = 0; r < 8; ++r) acc[i2][j2][r] = bv;
    }

  // staging coordinates (vector loads, no divergence)
  const int ar = tid >> 2;             // A row  (64 rows, 4 threads/row)
  const int ac = (tid & 3) * 16;       // A col base (16 halves / thread)
  const int wr = tid >> 1;             // W row  (128 rows, 2 threads/row)
  const int wc = (tid & 1) * 32;       // W col base (32 floats / thread)

  for (int k0 = 0; k0 < K; k0 += BK) {
    // ---- stage A tile (64 x 64) ----
    if (AF32) {
      const float4* g = (const float4*)&Af[(size_t)(m0 + ar) * K + k0 + ac];
      float4 f0 = g[0], f1 = g[1], f2 = g[2], f3 = g[3];
      v8h h0 = { (_Float16)f0.x, (_Float16)f0.y, (_Float16)f0.z, (_Float16)f0.w,
                 (_Float16)f1.x, (_Float16)f1.y, (_Float16)f1.z, (_Float16)f1.w };
      v8h h1 = { (_Float16)f2.x, (_Float16)f2.y, (_Float16)f2.z, (_Float16)f2.w,
                 (_Float16)f3.x, (_Float16)f3.y, (_Float16)f3.z, (_Float16)f3.w };
      *(v8h*)&As[ar][ac]     = h0;
      *(v8h*)&As[ar][ac + 8] = h1;
    } else {
      const v8h* g = (const v8h*)&Ah[(size_t)(m0 + ar) * K + k0 + ac];
      *(v8h*)&As[ar][ac]     = g[0];
      *(v8h*)&As[ar][ac + 8] = g[1];
    }
    // ---- stage W tile (128 x 64), f32 -> f16 ----
    {
      const float4* g = (const float4*)&W[(size_t)(n0 + wr) * K + k0 + wc];
      #pragma unroll
      for (int q = 0; q < 4; ++q) {
        float4 fa = g[2 * q], fb = g[2 * q + 1];
        v8h h = { (_Float16)fa.x, (_Float16)fa.y, (_Float16)fa.z, (_Float16)fa.w,
                  (_Float16)fb.x, (_Float16)fb.y, (_Float16)fb.z, (_Float16)fb.w };
        *(v8h*)&Ws[wr][wc + 8 * q] = h;
      }
    }
    // prefetch next K-stage (global_prefetch_b8)
    if (k0 + BK < K) {
      if (AF32) __builtin_prefetch(&Af[(size_t)(m0 + ar) * K + k0 + BK + ac], 0, 1);
      else      __builtin_prefetch(&Ah[(size_t)(m0 + ar) * K + k0 + BK + ac], 0, 1);
      __builtin_prefetch(&W[(size_t)(n0 + wr) * K + k0 + BK + wc], 0, 1);
    }
    __syncthreads();

    // ---- two WMMA k-chunks of 32 ----
    #pragma unroll
    for (int c = 0; c < 2; ++c) {
      const int c0 = c * 32;
      union { v16h v; v8h h2[2]; } a0, a1, b0, b1;
      {
        int row0 = wm + (lane & 15);
        int row1 = wm + 16 + (lane & 15);
        a0.h2[0] = *(const v8h*)&As[row0][c0 + 8 * half];
        a0.h2[1] = *(const v8h*)&As[row0][c0 + 16 + 8 * half];
        a1.h2[0] = *(const v8h*)&As[row1][c0 + 8 * half];
        a1.h2[1] = *(const v8h*)&As[row1][c0 + 16 + 8 * half];
      }
      {
        int col0 = wn + (lane & 15);
        int col1 = wn + 16 + (lane & 15);
        b0.h2[0] = *(const v8h*)&Ws[col0][c0 + 8 * half];
        b0.h2[1] = *(const v8h*)&Ws[col0][c0 + 16 + 8 * half];
        b1.h2[0] = *(const v8h*)&Ws[col1][c0 + 8 * half];
        b1.h2[1] = *(const v8h*)&Ws[col1][c0 + 16 + 8 * half];
      }
      acc[0][0] = __builtin_amdgcn_wmma_f32_16x16x32_f16(
          false, a0.v, false, b0.v, (short)0, acc[0][0], false, false);
      acc[0][1] = __builtin_amdgcn_wmma_f32_16x16x32_f16(
          false, a0.v, false, b1.v, (short)0, acc[0][1], false, false);
      acc[1][0] = __builtin_amdgcn_wmma_f32_16x16x32_f16(
          false, a1.v, false, b0.v, (short)0, acc[1][0], false, false);
      acc[1][1] = __builtin_amdgcn_wmma_f32_16x16x32_f16(
          false, a1.v, false, b1.v, (short)0, acc[1][1], false, false);
    }
    __syncthreads();
  }

  // ---- store C (layout: vgpr r -> M = r + 8*(lane>>4), lane&15 -> N) ----
  float*    Of = (float*)Optr;
  _Float16* Oh = (_Float16*)Optr;
  #pragma unroll
  for (int i2 = 0; i2 < 2; ++i2)
    #pragma unroll
    for (int j2 = 0; j2 < 2; ++j2) {
      const int ncol = n0 + wn + j2 * 16 + (lane & 15);
      #pragma unroll
      for (int r = 0; r < 8; ++r) {
        const int mrow = m0 + wm + i2 * 16 + r + 8 * half;
        float v = acc[i2][j2][r] * scale;
        if (RELU) v = fmaxf(v, 0.f);
        if (OF16) Oh[(size_t)mrow * ldOut + ncol] = (_Float16)v;
        else      Of[(size_t)mrow * ldOut + ncol] = v;
      }
    }
}

// =====================================================================
// K4: fused attention: per (n,h) block computes logits over s=0..255,
// softmax along s, and o[n,h,:] = sum_s attn[s]*v[s,n,h,:].
// =====================================================================
__global__ __launch_bounds__(256) void wca_attention(
    const float* __restrict__ q, const _Float16* __restrict__ kh,
    const _Float16* __restrict__ vh, float* __restrict__ o)
{
  __shared__ float qs[DH_];
  __shared__ float red[256];
  __shared__ float attnv[256];
  __shared__ float part[8][DH_];

  const int n = blockIdx.x / NH_;
  const int h = blockIdx.x % NH_;
  const int tid = threadIdx.x;

  if (tid < DH_) qs[tid] = q[n * D_ + h * DH_ + tid];
  __syncthreads();

  const _Float16* krow = kh + ((size_t)tid * N_ + n) * D_ + h * DH_;
  float dot = 0.f;
  #pragma unroll
  for (int d8 = 0; d8 < DH_ / 8; ++d8) {
    v8h kv = *(const v8h*)&krow[d8 * 8];
    #pragma unroll
    for (int e = 0; e < 8; ++e) dot += (float)kv[e] * qs[d8 * 8 + e];
  }

  red[tid] = dot; __syncthreads();
  for (int off = 128; off > 0; off >>= 1) {
    if (tid < off) red[tid] = fmaxf(red[tid], red[tid + off]);
    __syncthreads();
  }
  float mx = red[0]; __syncthreads();
  float e = __expf(dot - mx);
  red[tid] = e; __syncthreads();
  for (int off = 128; off > 0; off >>= 1) {
    if (tid < off) red[tid] += red[tid + off];
    __syncthreads();
  }
  float sm = red[0]; __syncthreads();
  attnv[tid] = e / sm;
  __syncthreads();

  const int d = tid & (DH_ - 1);
  const int chunk = tid >> 5;
  float acc = 0.f;
  for (int s = chunk * 32; s < chunk * 32 + 32; ++s)
    acc += attnv[s] * (float)vh[((size_t)s * N_ + n) * D_ + h * DH_ + d];
  part[chunk][d] = acc;
  __syncthreads();
  if (tid < DH_) {
    float t = 0.f;
    #pragma unroll
    for (int c = 0; c < 8; ++c) t += part[c][tid];
    o[n * D_ + h * DH_ + tid] = t;
  }
}

// =====================================================================
// K5: residual + LayerNorm -> slots_out
// =====================================================================
__global__ __launch_bounds__(256) void wca_resid_ln(
    const float* __restrict__ slots, const float* __restrict__ delta,
    const float* __restrict__ g, const float* __restrict__ b,
    float* __restrict__ outp)
{
  __shared__ float red[256];
  const int n = blockIdx.x, d = threadIdx.x;
  float x = slots[n * D_ + d] + delta[n * D_ + d];
  red[d] = x; __syncthreads();
  for (int off = 128; off > 0; off >>= 1) {
    if (d < off) red[d] += red[d + off];
    __syncthreads();
  }
  float mu = red[0] * (1.f / D_); __syncthreads();
  float c = x - mu;
  red[d] = c * c; __syncthreads();
  for (int off = 128; off > 0; off >>= 1) {
    if (d < off) red[d] += red[d + off];
    __syncthreads();
  }
  float var = red[0] * (1.f / D_);
  outp[n * D_ + d] = c * rsqrtf(var + EPS_) * g[d] + b[d];
}

// =====================================================================
// K6: broadcast proj_slots (N x 128, f16) into left half of proj_sf rows
// =====================================================================
__global__ __launch_bounds__(256) void wca_bcast_slots(
    const _Float16* __restrict__ ps, _Float16* __restrict__ psf)
{
  size_t idx = (size_t)blockIdx.x * blockDim.x + threadIdx.x; // S*N*128
  int j = (int)(idx & 127);
  size_t row = idx >> 7;
  int n = (int)(row % N_);
  psf[row * D_ + j] = ps[n * H2_ + j];
}

// =====================================================================
// K7: seg head final layer (Nout=3) + softmax -> seg_probs (f32)
// =====================================================================
__global__ __launch_bounds__(256) void wca_seg_final(
    const _Float16* __restrict__ hidden, const float* __restrict__ w2,
    const float* __restrict__ b2, float* __restrict__ probs)
{
  __shared__ float w2s[3 * D_];
  __shared__ float b2s[3];
  const int tid = threadIdx.x;
  for (int i = tid; i < 3 * D_; i += 256) w2s[i] = w2[i];
  if (tid < 3) b2s[tid] = b2[tid];
  __syncthreads();
  const size_t row = (size_t)blockIdx.x * 256 + tid;  // row = s*N + n
  const _Float16* hr = hidden + row * D_;
  float dots[3] = {0.f, 0.f, 0.f};
  for (int j8 = 0; j8 < D_ / 8; ++j8) {
    v8h hv = *(const v8h*)&hr[j8 * 8];
    #pragma unroll
    for (int e = 0; e < 8; ++e) {
      float hf = (float)hv[e];
      int j = j8 * 8 + e;
      dots[0] += hf * w2s[0 * D_ + j];
      dots[1] += hf * w2s[1 * D_ + j];
      dots[2] += hf * w2s[2 * D_ + j];
    }
  }
  dots[0] += b2s[0]; dots[1] += b2s[1]; dots[2] += b2s[2];
  float mx = fmaxf(dots[0], fmaxf(dots[1], dots[2]));
  float e0 = __expf(dots[0] - mx), e1 = __expf(dots[1] - mx), e2 = __expf(dots[2] - mx);
  float inv = 1.f / (e0 + e1 + e2);
  probs[row * 3 + 0] = e0 * inv;
  probs[row * 3 + 1] = e1 * inv;
  probs[row * 3 + 2] = e2 * inv;
}

// =====================================================================
// K8: seg_maps copy (float4 grid-stride) and K9: scatter seg_probs
// =====================================================================
__global__ __launch_bounds__(256) void wca_copy4(
    const float4* __restrict__ src, float4* __restrict__ dst, size_t n4)
{
  size_t i = (size_t)blockIdx.x * blockDim.x + threadIdx.x;
  size_t stride = (size_t)gridDim.x * blockDim.x;
  for (; i < n4; i += stride) dst[i] = src[i];
}

__global__ __launch_bounds__(256) void wca_scatter(
    const float* __restrict__ probs, const int* __restrict__ fidx,
    float* __restrict__ segout)
{
  const size_t row = (size_t)blockIdx.x * 256 + threadIdx.x; // s*N + n
  const int s = (int)(row / N_), n = (int)(row % N_);
  const int i = fidx[n * S_ + s];
  #pragma unroll
  for (int c = 0; c < 3; ++c)
    segout[(size_t)n * 3 * HW_ + (size_t)c * HW_ + i] = probs[row * 3 + c];
}

// =====================================================================
// host-side launch
// =====================================================================
static inline void launch_gemm(const void* A, bool aF32, const float* W,
                               const float* bias, void* O, bool oF16, int M,
                               int K, int Nout, int ldOut, float scale,
                               bool relu, hipStream_t st)
{
  dim3 g(Nout / BN, M / BM);
  dim3 blk(256);
  if (aF32) {
    if (oF16) {
      if (relu) wca_gemm_wmma<true, true, true><<<g, blk, 0, st>>>(A, W, bias, O, M, K, Nout, ldOut, scale);
      else      wca_gemm_wmma<true, true, false><<<g, blk, 0, st>>>(A, W, bias, O, M, K, Nout, ldOut, scale);
    } else {
      if (relu) wca_gemm_wmma<true, false, true><<<g, blk, 0, st>>>(A, W, bias, O, M, K, Nout, ldOut, scale);
      else      wca_gemm_wmma<true, false, false><<<g, blk, 0, st>>>(A, W, bias, O, M, K, Nout, ldOut, scale);
    }
  } else {
    if (oF16) {
      if (relu) wca_gemm_wmma<false, true, true><<<g, blk, 0, st>>>(A, W, bias, O, M, K, Nout, ldOut, scale);
      else      wca_gemm_wmma<false, true, false><<<g, blk, 0, st>>>(A, W, bias, O, M, K, Nout, ldOut, scale);
    } else {
      if (relu) wca_gemm_wmma<false, false, true><<<g, blk, 0, st>>>(A, W, bias, O, M, K, Nout, ldOut, scale);
      else      wca_gemm_wmma<false, false, false><<<g, blk, 0, st>>>(A, W, bias, O, M, K, Nout, ldOut, scale);
    }
  }
}

extern "C" void kernel_launch(void* const* d_in, const int* in_sizes, int n_in,
                              void* d_out, int out_size, void* d_ws, size_t ws_size,
                              hipStream_t stream)
{
  (void)in_sizes; (void)n_in; (void)out_size; (void)ws_size;

  const float* features   = (const float*)d_in[0];
  const float* pos_enc    = (const float*)d_in[1];
  const int*   batch_idx  = (const int*)d_in[2];
  const float* curio      = (const float*)d_in[3];
  const float* seg_maps   = (const float*)d_in[4];
  const float* slots      = (const float*)d_in[5];
  const int*   cov_sel    = (const int*)d_in[6];
  const float* in_proj_w  = (const float*)d_in[7];
  const float* in_proj_b  = (const float*)d_in[8];
  const float* out_proj_w = (const float*)d_in[9];
  const float* out_proj_b = (const float*)d_in[10];
  const float* ln_g       = (const float*)d_in[11];
  const float* ln_b       = (const float*)d_in[12];
  const float* slot_w1    = (const float*)d_in[13];
  const float* slot_b1    = (const float*)d_in[14];
  const float* slot_w2    = (const float*)d_in[15];
  const float* slot_b2    = (const float*)d_in[16];
  const float* feat_w1    = (const float*)d_in[17];
  const float* feat_b1    = (const float*)d_in[18];
  const float* feat_w2    = (const float*)d_in[19];
  const float* feat_b2    = (const float*)d_in[20];
  const float* seg_w1     = (const float*)d_in[21];
  const float* seg_b1     = (const float*)d_in[22];
  const float* seg_w2     = (const float*)d_in[23];
  const float* seg_b2     = (const float*)d_in[24];

  float* out_slots = (float*)d_out;                         // (1,N,D)
  float* out_seg   = out_slots + (size_t)N_ * D_;           // (N,3,HW)
  float* out_probs = out_seg + (size_t)N_ * 3 * HW_;        // (S,N,3)

  char* ws = (char*)d_ws;
  size_t cur = 0;
  auto alloc = [&](size_t bytes) -> void* {
    void* p = ws + cur;
    cur += (bytes + 255) & ~(size_t)255;
    return p;
  };
  const size_t SN = (size_t)S_ * N_;
  int*       feat_idx   = (int*)alloc(sizeof(int) * N_ * S_);
  _Float16*  keys_h     = (_Float16*)alloc(sizeof(_Float16) * SN * D_);
  _Float16*  values_h   = (_Float16*)alloc(sizeof(_Float16) * SN * D_);
  _Float16*  k_h        = (_Float16*)alloc(sizeof(_Float16) * SN * D_);
  _Float16*  v_h        = (_Float16*)alloc(sizeof(_Float16) * SN * D_);
  float*     q_f        = (float*)alloc(sizeof(float) * N_ * D_);
  float*     o_f        = (float*)alloc(sizeof(float) * N_ * D_);
  float*     delta_f    = (float*)alloc(sizeof(float) * N_ * D_);
  _Float16*  slot_h1    = (_Float16*)alloc(sizeof(_Float16) * N_ * H2_);
  _Float16*  proj_slots = (_Float16*)alloc(sizeof(_Float16) * N_ * H2_);
  _Float16*  feat_h1    = (_Float16*)alloc(sizeof(_Float16) * SN * H2_);
  _Float16*  proj_sf    = (_Float16*)alloc(sizeof(_Float16) * SN * D_);
  _Float16*  seg_hidden = (_Float16*)alloc(sizeof(_Float16) * SN * D_);

  // 1) per-row argsort + index selection (128KB dynamic LDS per WG)
  wca_sort_select<<<N_, 1024, HW_ * (sizeof(float) + sizeof(int)), stream>>>(
      curio, cov_sel, feat_idx);

  // 2) gather keys/values (f16)
  wca_gather<<<(int)SN, D_, 0, stream>>>(features, pos_enc, batch_idx,
                                         feat_idx, keys_h, values_h);

  // 3) projections (WMMA GEMMs)
  launch_gemm(keys_h,   false, in_proj_w + 1 * D_ * D_, in_proj_b + 1 * D_,
              k_h, true, (int)SN, D_, D_, D_, 1.f, false, stream);   // k
  launch_gemm(values_h, false, in_proj_w + 2 * D_ * D_, in_proj_b + 2 * D_,
              v_h, true, (int)SN, D_, D_, D_, 1.f, false, stream);   // v
  launch_gemm(slots,    true,  in_proj_w,               in_proj_b,
              q_f, false, N_, D_, D_, D_, QSCALE, false, stream);    // q (scaled)

  // 4) fused attention -> o
  wca_attention<<<N_ * NH_, 256, 0, stream>>>(q_f, k_h, v_h, o_f);

  // 5) out-proj, residual + LN -> slots_out
  launch_gemm(o_f, true, out_proj_w, out_proj_b, delta_f, false, N_, D_, D_,
              D_, 1.f, false, stream);
  wca_resid_ln<<<N_, D_, 0, stream>>>(slots, delta_f, ln_g, ln_b, out_slots);

  // 6) slot MLP (256 -> 128 relu -> 128)
  launch_gemm(out_slots, true, slot_w1, slot_b1, slot_h1, true, N_, D_, H2_,
              H2_, 1.f, true, stream);
  launch_gemm(slot_h1, false, slot_w2, slot_b2, proj_slots, true, N_, H2_,
              H2_, H2_, 1.f, false, stream);

  // 7) feat MLP (256 -> 128 relu -> 128); layer-2 writes right half of proj_sf
  launch_gemm(keys_h, false, feat_w1, feat_b1, feat_h1, true, (int)SN, D_,
              H2_, H2_, 1.f, true, stream);
  launch_gemm(feat_h1, false, feat_w2, feat_b2, (void*)(proj_sf + H2_), true,
              (int)SN, H2_, H2_, D_, 1.f, false, stream);

  // 8) broadcast proj_slots into left half of proj_sf
  wca_bcast_slots<<<(int)(SN * H2_ / 256), 256, 0, stream>>>(proj_slots,
                                                             proj_sf);

  // 9) seg MLP layer 1 (256 -> 256 relu), then final 3-way + softmax
  launch_gemm(proj_sf, false, seg_w1, seg_b1, seg_hidden, true, (int)SN, D_,
              D_, D_, 1.f, true, stream);
  wca_seg_final<<<(int)(SN / 256), 256, 0, stream>>>(seg_hidden, seg_w2,
                                                     seg_b2, out_probs);

  // 10) seg_maps copy + scatter of seg_probs
  const size_t n4 = (size_t)N_ * 3 * HW_ / 4;
  wca_copy4<<<8192, 256, 0, stream>>>((const float4*)seg_maps,
                                      (float4*)out_seg, n4);
  wca_scatter<<<(int)(SN / 256), 256, 0, stream>>>(out_probs, feat_idx,
                                                   out_seg);
}
